// Model_72335839199432
// MI455X (gfx1250) — compile-verified, matrix operand order checked by reading, not run on hardware
//
#include <hip/hip_runtime.h>
#include <math.h>

// ---- Model constants (from reference) ----
#define NB    8        // batch
#define LQ    1024     // seq len (enc) == label+pred (dec)
#define CIN   7
#define DM    512      // d_model
#define NH    8
#define DK    64       // d_model / n_heads
#define FFD   2048
#define UTOP  35       // 5 * ceil(ln 1024)
#define MROWS (NB*LQ)  // 8192 rows in every big GEMM

typedef _Float16 h16 __attribute__((ext_vector_type(16)));
typedef _Float16 h8  __attribute__((ext_vector_type(8)));
typedef _Float16 h2  __attribute__((ext_vector_type(2)));
typedef float    f32x8 __attribute__((ext_vector_type(8)));
typedef float    f2  __attribute__((ext_vector_type(2)));

__device__ __forceinline__ unsigned hash_u32(unsigned a) {
    a ^= a >> 16; a *= 0x7feb352dU;
    a ^= a >> 15; a *= 0x846ca68bU;
    a ^= a >> 16; return a;
}

// ============================================================================
// WMMA GEMM: C[M,N] = act(A[M,K] * W[K,N] + bias) [+ residual]
// Block = 256 threads = 8 waves; block tile 64(M) x 64(N); K step = 32.
// Waves arranged 4(M) x 2(N); each wave owns TWO 16x16 f32 accumulators
// (A fragment reused across both -> 2 v_wmma per wave per K step).
//
// LDS tiles are stored PRE-SWIZZLED in WMMA fragment order:
//   frag lane = (row&15) + 16*((k>>3)&1)
//   frag word v holds the fp16 pair (k, k+1) with v = ((k&7)>>1) | ((k>>4)<<2)
// so each lane's 16 fragment halves are 32 contiguous bytes -> fragment
// construction is exactly 2x ds_load_b128 per matrix.
//
// Software pipeline: global loads for tile k+1 are issued right after tile k
// is committed to LDS, so HBM/L2 latency overlaps the barrier + ds_load +
// v_wmma phase instead of stalling the staging phase on s_wait_loadcnt.
// ============================================================================
template<int ACT, int RES>   // ACT: 0 none, 1 exact GELU; RES: add residual
__global__ __launch_bounds__(256)
void gemm_wmma(const float* __restrict__ A, const float* __restrict__ W,
               const float* __restrict__ bias, const float* __restrict__ res,
               float* __restrict__ C, int M, int N, int K)
{
    __shared__ alignas(16) h2 Asw[4][32][8];   // [m 16-block][frag lane][word] 4KB
    __shared__ alignas(16) h2 Bsw[4][32][8];   // [n 16-block][frag lane][word] 4KB

    const int tid  = threadIdx.x;
    const int lane = tid & 31;                 // wave32
    const int wid  = tid >> 5;
    const int waveM = wid & 3;                 // 4 waves along M
    const int waveN = wid >> 2;                // 2 waves along N (each covers 32 cols)
    const int rowBase = blockIdx.y * 64;
    const int colBase = blockIdx.x * 64;

    // Per-thread staging assignment (fixed across K steps)
    int arow[4], acol[4], bn[4], bk[4];
    #pragma unroll
    for (int i = 0; i < 4; ++i) {
        int idx = tid + i * 256;
        arow[i] = idx >> 4;                    // 0..63
        acol[i] = (idx & 15) << 1;             // 0..30 (even)
        bn[i]   = idx & 63;
        bk[i]   = (idx >> 6) << 1;             // 0..30 (even)
    }

    f32x8 acc0 = {}, acc1 = {};
    f2    aReg[4];
    float bReg0[4], bReg1[4];

    // Prologue: load tile k0=0 into registers
    #pragma unroll
    for (int i = 0; i < 4; ++i) {
        aReg[i]  = *reinterpret_cast<const f2*>(&A[(size_t)(rowBase + arow[i]) * K + acol[i]]);
        bReg0[i] = W[(size_t)(bk[i])     * N + colBase + bn[i]];
        bReg1[i] = W[(size_t)(bk[i] + 1) * N + colBase + bn[i]];
    }

    for (int k0 = 0; k0 < K; k0 += 32) {
        // ---- Commit registers (tile k0) to swizzled LDS
        #pragma unroll
        for (int i = 0; i < 4; ++i) {
            int r = arow[i], c = acol[i];
            h2 pa; pa[0] = (_Float16)aReg[i][0]; pa[1] = (_Float16)aReg[i][1];
            Asw[r >> 4][(r & 15) + (((c >> 3) & 1) << 4)][((c & 7) >> 1) | ((c >> 4) << 2)] = pa;
            int n = bn[i], k = bk[i];
            h2 pb; pb[0] = (_Float16)bReg0[i]; pb[1] = (_Float16)bReg1[i];
            Bsw[n >> 4][(n & 15) + (((k >> 3) & 1) << 4)][((k & 7) >> 1) | ((k >> 4) << 2)] = pb;
        }
        // ---- Issue global loads for tile k0+32 (latency hidden behind WMMA phase)
        if (k0 + 32 < K) {
            #pragma unroll
            for (int i = 0; i < 4; ++i) {
                aReg[i]  = *reinterpret_cast<const f2*>(
                               &A[(size_t)(rowBase + arow[i]) * K + k0 + 32 + acol[i]]);
                bReg0[i] = W[(size_t)(k0 + 32 + bk[i])     * N + colBase + bn[i]];
                bReg1[i] = W[(size_t)(k0 + 32 + bk[i] + 1) * N + colBase + bn[i]];
            }
        }
        __syncthreads();

        // ---- Fragments: 2x ds_load_b128 each
        const h8* ap  = reinterpret_cast<const h8*>(&Asw[waveM][lane][0]);
        const h8* bp0 = reinterpret_cast<const h8*>(&Bsw[waveN * 2][lane][0]);
        const h8* bp1 = reinterpret_cast<const h8*>(&Bsw[waveN * 2 + 1][lane][0]);
        h16 af = __builtin_shufflevector(ap[0],  ap[1],  0,1,2,3,4,5,6,7,8,9,10,11,12,13,14,15);
        h16 b0 = __builtin_shufflevector(bp0[0], bp0[1], 0,1,2,3,4,5,6,7,8,9,10,11,12,13,14,15);
        h16 b1 = __builtin_shufflevector(bp1[0], bp1[1], 0,1,2,3,4,5,6,7,8,9,10,11,12,13,14,15);
        acc0 = __builtin_amdgcn_wmma_f32_16x16x32_f16(false, af, false, b0, (short)0, acc0, false, false);
        acc1 = __builtin_amdgcn_wmma_f32_16x16x32_f16(false, af, false, b1, (short)0, acc1, false, false);
        __syncthreads();
    }

    // C/D layout: lanes 0-15 hold rows m=0..7 in VGPR 0..7, lanes 16-31 rows 8..15
    const int n0 = colBase + waveN * 32 + (lane & 15);
    const int mb = rowBase + waveM * 16 + ((lane >> 4) << 3);
    const float bv0 = bias ? bias[n0] : 0.f;
    const float bv1 = bias ? bias[n0 + 16] : 0.f;
    #pragma unroll
    for (int i = 0; i < 8; ++i) {
        int row = mb + i;
        float v0 = acc0[i] + bv0;
        float v1 = acc1[i] + bv1;
        if (ACT == 1) {
            v0 = 0.5f * v0 * (1.f + erff(v0 * 0.70710678118f));
            v1 = 0.5f * v1 * (1.f + erff(v1 * 0.70710678118f));
        }
        if (RES) {
            v0 += res[(size_t)row * N + n0];
            v1 += res[(size_t)row * N + n0 + 16];
        }
        C[(size_t)row * N + n0]      = v0;
        C[(size_t)row * N + n0 + 16] = v1;
    }
}

// ============================================================================
// LayerNorm over last dim (512). One wave per row, 16 elems/lane.
// ============================================================================
__global__ __launch_bounds__(256)
void layernorm_k(const float* __restrict__ X, const float* __restrict__ g,
                 const float* __restrict__ b, float* __restrict__ Y, int rows)
{
    int wid = threadIdx.x >> 5, lane = threadIdx.x & 31;
    int row = blockIdx.x * 8 + wid;
    if (row >= rows) return;
    const float* xr = X + (size_t)row * DM;
    float vals[16], s = 0.f;
    #pragma unroll
    for (int i = 0; i < 16; ++i) { vals[i] = xr[lane + i * 32]; s += vals[i]; }
    #pragma unroll
    for (int o = 16; o > 0; o >>= 1) s += __shfl_xor(s, o, 32);
    float mean = s * (1.f / DM);
    float vs = 0.f;
    #pragma unroll
    for (int i = 0; i < 16; ++i) { float d = vals[i] - mean; vs += d * d; }
    #pragma unroll
    for (int o = 16; o > 0; o >>= 1) vs += __shfl_xor(vs, o, 32);
    float rstd = rsqrtf(vs * (1.f / DM) + 1e-5f);
    float* yr = Y + (size_t)row * DM;
    #pragma unroll
    for (int i = 0; i < 16; ++i) {
        int d = lane + i * 32;
        yr[d] = (vals[i] - mean) * rstd * g[d] + b[d];
    }
}

// ============================================================================
// Token embed (circular conv width 3 over CIN) + temporal mark + pos emb.
// mode 0: encoder mark l/L.  mode 1: decoder mark (label|pred schedule).
// ============================================================================
__global__ void embed_k(const float* __restrict__ xin, const float* __restrict__ Wtok,
                        const float* __restrict__ temp, float* __restrict__ out, int mode)
{
    int idx = blockIdx.x * blockDim.x + threadIdx.x;
    if (idx >= NB * LQ * DM) return;
    int d = idx % DM, l = (idx / DM) % LQ, b = idx / (DM * LQ);
    float acc = 0.f;
    #pragma unroll
    for (int w = 0; w < 3; ++w) {
        int ll = l + w - 1;
        ll = (ll < 0) ? (LQ - 1) : (ll >= LQ ? 0 : ll);
        const float* xr = xin + ((size_t)b * LQ + ll) * CIN;
        const float* wr = Wtok + (size_t)w * CIN * DM + d;
        #pragma unroll
        for (int c = 0; c < CIN; ++c) acc += xr[c] * wr[c * DM];
    }
    float mark = (mode == 0) ? ((float)l / (float)LQ)
               : (l < 512 ? (float)(512 + l) / 1024.f : (float)(l - 512) / 512.f);
    acc += mark * temp[d];
    int i2 = d >> 1;
    float freq = __expf(-(float)(2 * i2) * (9.210340371976184f / 512.f)); // ln(1e4)
    float ang = (float)l * freq;
    acc += (d & 1) ? __cosf(ang) : __sinf(ang);
    out[idx] = acc;
}

__global__ void xdec_prep(const float* __restrict__ x, float* __restrict__ xdec)
{
    int idx = blockIdx.x * blockDim.x + threadIdx.x;
    if (idx >= NB * LQ * CIN) return;
    int c = idx % CIN, j = (idx / CIN) % LQ, b = idx / (CIN * LQ);
    xdec[idx] = (j < 512) ? x[((size_t)b * LQ + 512 + j) * CIN + c] : 0.f;
}

// ============================================================================
// ProbSparse: sampled sparsity measure M = max(QK_s) - mean(QK_s)
// ============================================================================
__global__ void mscore_k(const float* __restrict__ q, const float* __restrict__ k,
                         float* __restrict__ msc, int seed, int LK)
{
    int idx = blockIdx.x * blockDim.x + threadIdx.x;   // (b,h,qi)
    if (idx >= NB * NH * LQ) return;
    int qi = idx % LQ, hh = (idx / LQ) % NH, b = idx / (LQ * NH);
    const float* qp = q + ((size_t)b * LQ + qi) * DM + hh * DK;
    float mx = -1e30f, sum = 0.f;
    for (int s = 0; s < UTOP; ++s) {
        unsigned r = hash_u32((unsigned)seed * 0x9E3779B9u ^ (qi * 0x85ebca6bu) ^ (s * 0xc2b2ae35u));
        int ki = (int)(r % (unsigned)LK);
        const float* kp = k + ((size_t)b * LQ + ki) * DM + hh * DK;
        float dot = 0.f;
        #pragma unroll
        for (int d = 0; d < DK; ++d) dot += qp[d] * kp[d];
        mx = fmaxf(mx, dot); sum += dot;
    }
    msc[idx] = mx - sum / (float)LK;
}

// Top-U selection per (b,h): 35 rounds of parallel argmax
__global__ __launch_bounds__(256)
void topu_k(const float* __restrict__ msc, int* __restrict__ mtop)
{
    __shared__ float vals[LQ];
    __shared__ float rv[256];
    __shared__ int   ri[256];
    int bh = blockIdx.x, tid = threadIdx.x;
    const float* mp = msc + (size_t)bh * LQ;
    for (int i = tid; i < LQ; i += 256) vals[i] = mp[i];
    __syncthreads();
    for (int t = 0; t < UTOP; ++t) {
        float bestv = -1e30f; int besti = 0;
        for (int i = tid; i < LQ; i += 256)
            if (vals[i] > bestv) { bestv = vals[i]; besti = i; }
        rv[tid] = bestv; ri[tid] = besti;
        __syncthreads();
        for (int o = 128; o > 0; o >>= 1) {
            if (tid < o && rv[tid + o] > rv[tid]) { rv[tid] = rv[tid + o]; ri[tid] = ri[tid + o]; }
            __syncthreads();
        }
        if (tid == 0) { mtop[bh * UTOP + t] = ri[0]; vals[ri[0]] = -1e30f; }
        __syncthreads();
    }
}

// Non-causal context: broadcast mean(V) over all queries
__global__ __launch_bounds__(256)
void ctxmean_k(const float* __restrict__ v, float* __restrict__ ctx)
{
    __shared__ float mv[DK];
    int bh = blockIdx.x, b = bh / NH, hh = bh % NH, tid = threadIdx.x;
    if (tid < DK) {
        float s = 0.f;
        for (int l = 0; l < LQ; ++l) s += v[((size_t)b * LQ + l) * DM + hh * DK + tid];
        mv[tid] = s * (1.f / LQ);
    }
    __syncthreads();
    int d = tid & 63, q0 = tid >> 6;
    for (int qq = q0; qq < LQ; qq += 4)
        ctx[((size_t)b * LQ + qq) * DM + hh * DK + d] = mv[d];
}

// Causal context: cumulative sum of V over key axis
__global__ void ctxcum_k(const float* __restrict__ v, float* __restrict__ ctx)
{
    int idx = blockIdx.x * blockDim.x + threadIdx.x;   // (b,h,d)
    if (idx >= NB * NH * DK) return;
    int d = idx % DK, hh = (idx / DK) % NH, b = idx / (DK * NH);
    float acc = 0.f;
    for (int l = 0; l < LQ; ++l) {
        size_t o = ((size_t)b * LQ + l) * DM + hh * DK + d;
        acc += v[o]; ctx[o] = acc;
    }
}

// Full attention for the UTOP selected queries; scatter into ctx rows
__global__ __launch_bounds__(256)
void attnupd_k(const float* __restrict__ q, const float* __restrict__ k,
               const float* __restrict__ v, const int* __restrict__ mtop,
               float* __restrict__ ctx, int causal)
{
    __shared__ float qv[DK];
    __shared__ float sc[LQ];
    __shared__ float red[256];
    __shared__ float part[4][DK];
    int u = blockIdx.x % UTOP, bh = blockIdx.x / UTOP;
    int b = bh / NH, hh = bh % NH, tid = threadIdx.x;
    int qidx = mtop[bh * UTOP + u];
    if (tid < DK) qv[tid] = q[((size_t)b * LQ + qidx) * DM + hh * DK + tid];
    __syncthreads();

    float lmax = -1e30f;
    for (int kk = tid; kk < LQ; kk += 256) {
        const float* kp = k + ((size_t)b * LQ + kk) * DM + hh * DK;
        float dot = 0.f;
        #pragma unroll
        for (int d = 0; d < DK; ++d) dot += qv[d] * kp[d];
        dot *= 0.125f;                                  // 1/sqrt(64)
        if (causal && kk > qidx) dot = -1e30f;
        sc[kk] = dot;
        lmax = fmaxf(lmax, dot);
    }
    red[tid] = lmax; __syncthreads();
    for (int o = 128; o > 0; o >>= 1) { if (tid < o) red[tid] = fmaxf(red[tid], red[tid + o]); __syncthreads(); }
    float mx = red[0]; __syncthreads();

    float lsum = 0.f;
    for (int kk = tid; kk < LQ; kk += 256) { float e = __expf(sc[kk] - mx); sc[kk] = e; lsum += e; }
    red[tid] = lsum; __syncthreads();
    for (int o = 128; o > 0; o >>= 1) { if (tid < o) red[tid] += red[tid + o]; __syncthreads(); }
    float inv = 1.f / red[0];

    int d = tid & 63, g = tid >> 6;
    float a = 0.f;
    for (int kk = g * 256; kk < g * 256 + 256; ++kk)
        a += sc[kk] * v[((size_t)b * LQ + kk) * DM + hh * DK + d];
    part[g][d] = a; __syncthreads();
    if (g == 0) {
        float t = (part[0][d] + part[1][d] + part[2][d] + part[3][d]) * inv;
        ctx[((size_t)b * LQ + qidx) * DM + hh * DK + d] = t;
    }
}

// Final projection d_model -> CIN on the last PRED_LEN rows
__global__ void proj_k(const float* __restrict__ dn, const float* __restrict__ W,
                       const float* __restrict__ bias, float* __restrict__ out)
{
    int idx = blockIdx.x * blockDim.x + threadIdx.x;   // (b, j, c)
    if (idx >= NB * 512 * CIN) return;
    int c = idx % CIN, j = (idx / CIN) % 512, b = idx / (CIN * 512);
    const float* dr = dn + ((size_t)b * LQ + 512 + j) * DM;
    float acc = bias[c];
    for (int d = 0; d < DM; ++d) acc += dr[d] * W[d * CIN + c];
    out[idx] = acc;
}

// ============================================================================
// Orchestration
// ============================================================================
extern "C" void kernel_launch(void* const* d_in, const int* in_sizes, int n_in,
                              void* d_out, int out_size, void* d_ws, size_t ws_size,
                              hipStream_t stream)
{
    const float* P[96];
    for (int i = 0; i < n_in && i < 96; ++i) P[i] = (const float*)d_in[i];

    // Flattened param order (setup_inputs dict insertion order):
    // 0:x 1:enc_tok 2:enc_temp 3:dec_tok 4:dec_temp
    // 5..36: enc_layers[2] x {qW,qb,kW,kb,vW,vb,oW,ob,ff1W,ff1b,ff2W,ff2b,ln1g,ln1b,ln2g,ln2b}
    // 37,38: enc_norm g,b
    // 39..64: dec layer {self x8, cross x8, ff1W,ff1b,ff2W,ff2b,ln1g,ln1b,ln2g,ln2b,ln3g,ln3b}
    // 65,66: dec_norm g,b   67,68: proj W,b
    const float* x = P[0];

    // Workspace carve-out
    float* ws = (float*)d_ws;
    size_t o = 0;
    auto alloc = [&](size_t n) { float* p = ws + o; o += n; return p; };
    float* hbuf = alloc((size_t)MROWS * DM);
    float* tmp  = alloc((size_t)MROWS * DM);
    float* qb   = alloc((size_t)MROWS * DM);
    float* kb   = alloc((size_t)MROWS * DM);
    float* vb   = alloc((size_t)MROWS * DM);
    float* ctx  = alloc((size_t)MROWS * DM);
    float* ffb  = alloc((size_t)MROWS * FFD);
    float* enc  = alloc((size_t)MROWS * DM);
    float* dbuf = alloc((size_t)MROWS * DM);
    float* xdec = alloc((size_t)NB * LQ * CIN);
    float* msc  = alloc((size_t)NB * NH * LQ);
    int*   mtop = (int*)alloc((size_t)NB * NH * UTOP);

    auto gemm = [&](const float* A, const float* W, const float* bias,
                    const float* res, float* C, int N, int K, int act) {
        dim3 grid(N / 64, MROWS / 64);
        if (act)      gemm_wmma<1, 0><<<grid, 256, 0, stream>>>(A, W, bias, res, C, MROWS, N, K);
        else if (res) gemm_wmma<0, 1><<<grid, 256, 0, stream>>>(A, W, bias, res, C, MROWS, N, K);
        else          gemm_wmma<0, 0><<<grid, 256, 0, stream>>>(A, W, bias, res, C, MROWS, N, K);
    };
    auto lnorm = [&](const float* X, const float* g, const float* b, float* Y) {
        layernorm_k<<<MROWS / 8, 256, 0, stream>>>(X, g, b, Y, MROWS);
    };
    // ProbSparse attention: Q from xq, K/V from xkv, output ctx (b,l,h,dk)
    auto prob_attn = [&](const float* xq, const float* xkv, const float* const* w,
                         int causal, int seed) {
        gemm(xq,  w[0], w[1], nullptr, qb, DM, DM, 0);
        gemm(xkv, w[2], w[3], nullptr, kb, DM, DM, 0);
        gemm(xkv, w[4], w[5], nullptr, vb, DM, DM, 0);
        mscore_k<<<(NB * NH * LQ + 255) / 256, 256, 0, stream>>>(qb, kb, msc, seed, LQ);
        topu_k<<<NB * NH, 256, 0, stream>>>(msc, mtop);
        if (causal) ctxcum_k<<<(NB * NH * DK + 255) / 256, 256, 0, stream>>>(vb, ctx);
        else        ctxmean_k<<<NB * NH, 256, 0, stream>>>(vb, ctx);
        attnupd_k<<<NB * NH * UTOP, 256, 0, stream>>>(qb, kb, vb, mtop, ctx, causal);
    };

    // ---------------- Encoder ----------------
    embed_k<<<(NB * LQ * DM + 255) / 256, 256, 0, stream>>>(x, P[1], P[2], hbuf, 0);
    for (int l = 0; l < 2; ++l) {
        const int base = 5 + 16 * l;
        const float* aw[6] = { P[base], P[base+1], P[base+2], P[base+3], P[base+4], P[base+5] };
        prob_attn(hbuf, hbuf, aw, 0, /*seed c=*/l);
        gemm(ctx, P[base+6], P[base+7], hbuf, tmp, DM, DM, 0);        // +residual
        lnorm(tmp, P[base+12], P[base+13], hbuf);                     // ln1
        gemm(hbuf, P[base+8], P[base+9], nullptr, ffb, FFD, DM, 1);   // ff1 + GELU
        gemm(ffb, P[base+10], P[base+11], hbuf, tmp, DM, FFD, 0);     // ff2 + residual
        lnorm(tmp, P[base+14], P[base+15], hbuf);                     // ln2
    }
    lnorm(hbuf, P[37], P[38], enc);                                   // enc_norm

    // ---------------- Decoder ----------------
    xdec_prep<<<(NB * LQ * CIN + 255) / 256, 256, 0, stream>>>(x, xdec);
    embed_k<<<(NB * LQ * DM + 255) / 256, 256, 0, stream>>>(xdec, P[3], P[4], dbuf, 1);

    {
        // self-attention (causal), weights 39..46
        const float* sw[6] = { P[39], P[40], P[41], P[42], P[43], P[44] };
        prob_attn(dbuf, dbuf, sw, 1, /*seed c=*/2);
        gemm(ctx, P[45], P[46], dbuf, tmp, DM, DM, 0);
        lnorm(tmp, P[59], P[60], dbuf);                               // ln1

        // cross-attention (non-causal), weights 47..54, K/V from encoder
        const float* cw[6] = { P[47], P[48], P[49], P[50], P[51], P[52] };
        prob_attn(dbuf, enc, cw, 0, /*seed c=*/3);
        gemm(ctx, P[53], P[54], dbuf, tmp, DM, DM, 0);                // d + cross
        lnorm(tmp, P[61], P[62], dbuf);                               // ln2

        gemm(dbuf, P[55], P[56], nullptr, ffb, FFD, DM, 1);           // ff1 + GELU
        gemm(ffb, P[57], P[58], dbuf, tmp, DM, FFD, 0);               // ff2 + residual
        lnorm(tmp, P[63], P[64], dbuf);                               // ln3
    }
    lnorm(dbuf, P[65], P[66], tmp);                                   // dec_norm
    proj_k<<<(NB * 512 * CIN + 255) / 256, 256, 0, stream>>>(tmp, P[67], P[68], (float*)d_out);
}
